// DenseGATLayer_51668456571273
// MI455X (gfx1250) — matrix operand order repaired and checked by reading, not compiled, not run on hardware
//
#include <hip/hip_runtime.h>
#include <stdint.h>

#define B_ 4
#define N_ 2048
#define IN_ 128
#define O_ 32
#define H_ 8
#define NEG_SLOPE 0.2f

typedef __attribute__((ext_vector_type(16))) _Float16     v16h;
typedef __attribute__((ext_vector_type(8)))  _Float16     v8h;
typedef __attribute__((ext_vector_type(8)))  float        v8f;
typedef __attribute__((ext_vector_type(4)))  unsigned int u32x4;
typedef __attribute__((ext_vector_type(4)))  int          i32x4;
typedef __attribute__((ext_vector_type(8)))  int          i32x8;

// 16-bit A-matrix 16x32 (MxK) lane/element -> K mapping (ISA 7.12.2)
__device__ __forceinline__ int kmapA(int t, int hi) { return t + 8 * ((t >> 3) + hi); }

#define HAS_TDM (__has_builtin(__builtin_amdgcn_tensor_load_to_lds) && \
                 __has_builtin(__builtin_amdgcn_s_wait_tensorcnt))

#if HAS_TDM
// Issue one TDM descriptor: 16x32 f32 tile (row stride N_) global -> LDS.
__device__ __forceinline__ void tdm_load_tile(const float* gsrc, const float* ldsDst) {
  const unsigned long long ga = (unsigned long long)(uintptr_t)gsrc;
  u32x4 g0;
  g0[0] = 1u;                                                    // count=1 descriptor
  g0[1] = (unsigned int)(uintptr_t)ldsDst;                       // lds_addr
  g0[2] = (unsigned int)ga;                                      // gaddr[31:0]
  g0[3] = (unsigned int)((ga >> 32) & 0x01FFFFFFu) | (2u << 30); // gaddr[56:32] | type=2
  i32x8 g1;
  g1[0] = 2 << 16;                                               // data_size = 4B
  g1[1] = (int)((unsigned)(N_ & 0xFFFF) << 16);                  // tensor_dim0 lo16
  g1[2] = (int)(((unsigned)(N_ & 0xFFFF) << 16) | ((unsigned)N_ >> 16)); // tdim1 lo|tdim0 hi
  g1[3] = (int)((32u << 16) | ((unsigned)N_ >> 16));             // tile_dim0=32 | tdim1 hi
  g1[4] = 16;                                                    // tile_dim1=16, tile_dim2=0
  g1[5] = N_;                                                    // tensor_dim0_stride lo32
  g1[6] = 0; g1[7] = 0;
  i32x4 gz4 = {0, 0, 0, 0};
  i32x8 gz8 = {0, 0, 0, 0, 0, 0, 0, 0};
  __builtin_amdgcn_tensor_load_to_lds(g0, g1, gz4, gz4, gz8, 0);
}
#endif

// ---------------------------------------------------------------------------
// Kernel 0: WT[h][o][k] = (f16) W[h][k][o]   (K-contiguous B operand for k_wh)
// ---------------------------------------------------------------------------
__global__ __launch_bounds__(256) void k_wt(const float* __restrict__ W,
                                            _Float16* __restrict__ WT) {
  const int idx = blockIdx.x * 256 + threadIdx.x;    // (h*O + o)*IN + k
  const int k = idx & (IN_ - 1);
  const int ho = idx >> 7;                           // h*O + o
  const int o = ho & (O_ - 1);
  const int h = ho >> 5;
  WT[idx] = (_Float16)W[((size_t)h * IN_ + k) * O_ + o];
}

// ---------------------------------------------------------------------------
// Kernel 1: WhT[bh][o][i] = (x[b] @ W[h])^T  via WMMA f16 (f32 accum)
// grid = B*H*(N/128), block = 256 (8 waves, wave w owns 16 rows)
// ---------------------------------------------------------------------------
__global__ __launch_bounds__(256) void k_wh(const float* __restrict__ x,
                                            const _Float16* __restrict__ WT,
                                            _Float16* __restrict__ WhT) {
  const int lane = threadIdx.x & 31;
  const int wave = threadIdx.x >> 5;
  const int hi = lane >> 4;
  const int lo = lane & 15;
  const int itile = blockIdx.x & 15;   // N/128 = 16 tiles of 128 rows
  const int bh = blockIdx.x >> 4;      // b*H + h
  const int h = bh & 7;
  const int b = bh >> 3;
  const int i0 = itile * 128 + wave * 16;

  v8f c0 = {}; v8f c1 = {};
  const float* xrow = x + ((size_t)(b * N_ + i0 + lo)) * IN_;

#pragma unroll
  for (int kc = 0; kc < 4; ++kc) {
    const int kb = kc * 32;
    v16h A;
#pragma unroll
    for (int t = 0; t < 16; ++t) A[t] = (_Float16)xrow[kb + kmapA(t, hi)];
    // B operand: two 8-half runs per lane at K = 8*hi and 16 + 8*hi (K-major WT)
    const _Float16* wb0 = WT + ((size_t)h * O_ + lo) * IN_ + kb;
    const _Float16* wb1 = WT + ((size_t)h * O_ + 16 + lo) * IN_ + kb;
    const v16h B0 = __builtin_shufflevector(*(const v8h*)(wb0 + 8 * hi), *(const v8h*)(wb0 + 16 + 8 * hi), 0, 1, 2, 3, 4, 5, 6, 7, 8, 9, 10, 11, 12, 13, 14, 15);
    const v16h B1 = __builtin_shufflevector(*(const v8h*)(wb1 + 8 * hi), *(const v8h*)(wb1 + 16 + 8 * hi), 0, 1, 2, 3, 4, 5, 6, 7, 8, 9, 10, 11, 12, 13, 14, 15);
    c0 = __builtin_amdgcn_wmma_f32_16x16x32_f16(false, A, false, B0, (short)0, c0, false, false);
    c1 = __builtin_amdgcn_wmma_f32_16x16x32_f16(false, A, false, B1, (short)0, c1, false, false);
  }

  // C layout: lane holds column lo (c0) / 16+lo (c1), rows i0 + r + 8*hi.
  // Store transposed (K-contiguous for the attention GEMM): one b128 per column.
  v8h s0, s1;
#pragma unroll
  for (int r = 0; r < 8; ++r) { s0[r] = (_Float16)c0[r]; s1[r] = (_Float16)c1[r]; }
  *(v8h*)(WhT + ((size_t)bh * O_ + lo) * N_ + i0 + 8 * hi)      = s0;
  *(v8h*)(WhT + ((size_t)bh * O_ + 16 + lo) * N_ + i0 + 8 * hi) = s1;
}

// ---------------------------------------------------------------------------
// Kernel 2: alpha_src/alpha_dst = Wh . a_src / a_dst  (coalesced over i)
// ---------------------------------------------------------------------------
__global__ __launch_bounds__(256) void k_alpha(const _Float16* __restrict__ WhT,
                                               const float* __restrict__ a_src,
                                               const float* __restrict__ a_dst,
                                               float* __restrict__ aS,
                                               float* __restrict__ aD) {
  const int idx = blockIdx.x * 256 + threadIdx.x;  // (b*H+h)*N + i
  const int i = idx & (N_ - 1);
  const int bh = idx >> 11;
  const int h = bh & 7;
  float s = 0.f, d = 0.f;
#pragma unroll
  for (int o = 0; o < O_; ++o) {
    const float wv = (float)WhT[((size_t)bh * O_ + o) * N_ + i];
    s += wv * a_src[h * O_ + o];
    d += wv * a_dst[h * O_ + o];
  }
  aS[idx] = s;
  aD[idx] = d;
}

// ---------------------------------------------------------------------------
// Kernel 3: per-row online softmax stats (m, 1/l). One wave per (b,i,h);
// 8 heads of one (b,i) share a block so the adj row stays cache-hot.
// float4 streams over adj / alpha_dst.
// ---------------------------------------------------------------------------
__global__ __launch_bounds__(256) void k_stats(const float* __restrict__ adj,
                                               const float* __restrict__ aS,
                                               const float* __restrict__ aD,
                                               float* __restrict__ mrow,
                                               float* __restrict__ invl) {
  const int gw = (blockIdx.x * 256 + threadIdx.x) >> 5;  // (b*N + i)*H + h
  const int lane = threadIdx.x & 31;
  const int h = gw & 7;
  const int i = (gw >> 3) & (N_ - 1);
  const int b = gw >> 14;
  const int bh = b * H_ + h;

  const float sA = aS[(size_t)bh * N_ + i];
  const float4* arow4 = (const float4*)(adj + ((size_t)b * N_ + i) * N_);
  const float4* dvec4 = (const float4*)(aD + (size_t)bh * N_);

  float m = -1e30f, l = 0.f;
  for (int jb = 0; jb < N_ / 128; ++jb) {
    const int j4 = jb * 32 + lane;
    const float4 av = arow4[j4];
    const float4 dv = dvec4[j4];
    const int j = j4 * 4;
    const float a[4] = {av.x, av.y, av.z, av.w};
    const float dd[4] = {dv.x, dv.y, dv.z, dv.w};
#pragma unroll
    for (int u = 0; u < 4; ++u) {
      if (a[u] > 0.f || (j + u) == i) {          // adj + I mask
        float e = sA + dd[u];
        e = (e >= 0.f) ? e : NEG_SLOPE * e;
        const float mN = fmaxf(m, e);
        l = l * __expf(m - mN) + __expf(e - mN);
        m = mN;
      }
    }
  }
#pragma unroll
  for (int off = 16; off > 0; off >>= 1) {
    const float m2 = __shfl_xor(m, off, 32);
    const float l2 = __shfl_xor(l, off, 32);
    const float mN = fmaxf(m, m2);
    l = l * __expf(m - mN) + l2 * __expf(m2 - mN);
    m = mN;
  }
  if (lane == 0) {
    mrow[(size_t)bh * N_ + i] = m;
    invl[(size_t)bh * N_ + i] = 1.f / l;
  }
}

// ---------------------------------------------------------------------------
// Kernel 4: out = relu( softmax(e) @ Wh ).  Block = (b, 16-row tile),
// wave = head. adj tile TDM-staged to LDS (double-buffered), shared by 8 heads.
// ---------------------------------------------------------------------------
__global__ __launch_bounds__(256) void k_attn_out(const float* __restrict__ adj,
                                                  const _Float16* __restrict__ WhT,
                                                  const float* __restrict__ aS,
                                                  const float* __restrict__ aD,
                                                  const float* __restrict__ mrow,
                                                  const float* __restrict__ invl,
                                                  float* __restrict__ out) {
  const int lane = threadIdx.x & 31;
  const int h = threadIdx.x >> 5;     // wave index == head
  const int hi = lane >> 4;
  const int lo = lane & 15;
  const int b = blockIdx.x >> 7;      // N/16 = 128 row tiles
  const int i0 = (blockIdx.x & 127) * 16;
  const int bh = b * H_ + h;

  __shared__ __align__(16) float adjTile[2][16 * 32];

  const int irow = i0 + lo;           // p-fragment: this lane's row
  const float sA = aS[(size_t)bh * N_ + irow];
  const float mI = mrow[(size_t)bh * N_ + irow];
  const float* dvec = aD + (size_t)bh * N_;
  const float* atile = adj + ((size_t)b * N_ + i0) * N_;
  const _Float16* col0 = WhT + ((size_t)bh * O_ + lo) * N_;       // B0 column
  const _Float16* col1 = WhT + ((size_t)bh * O_ + 16 + lo) * N_;  // B1 column

  v8f c0 = {}; v8f c1 = {};

#if HAS_TDM
  if (threadIdx.x < 32) tdm_load_tile(atile, &adjTile[0][0]);
#endif

#pragma unroll 2
  for (int jc = 0; jc < N_ / 32; ++jc) {
    const int j0 = jc * 32;
    const int cur = jc & 1;
#if HAS_TDM
    if (threadIdx.x < 32) __builtin_amdgcn_s_wait_tensorcnt((unsigned short)0);
    __syncthreads();                  // tile jc ready; prev buffer free
    if (threadIdx.x < 32 && jc + 1 < N_ / 32)
      tdm_load_tile(atile + j0 + 32, &adjTile[cur ^ 1][0]);
#else
    __syncthreads();
    for (int t = threadIdx.x; t < 16 * 32; t += 256)
      adjTile[cur][t] = atile[((size_t)(t >> 5)) * N_ + j0 + (t & 31)];
    __syncthreads();
#endif

    // p fragment: two contiguous 8-float runs per lane (K = t + 8*((t>>3)+hi))
    float av[16], dvv[16];
    const float* trow = &adjTile[cur][lo * 32];
    *(float4*)&av[0]   = *(const float4*)&trow[8 * hi];
    *(float4*)&av[4]   = *(const float4*)&trow[8 * hi + 4];
    *(float4*)&av[8]   = *(const float4*)&trow[16 + 8 * hi];
    *(float4*)&av[12]  = *(const float4*)&trow[16 + 8 * hi + 4];
    *(float4*)&dvv[0]  = *(const float4*)&dvec[j0 + 8 * hi];
    *(float4*)&dvv[4]  = *(const float4*)&dvec[j0 + 8 * hi + 4];
    *(float4*)&dvv[8]  = *(const float4*)&dvec[j0 + 16 + 8 * hi];
    *(float4*)&dvv[12] = *(const float4*)&dvec[j0 + 16 + 8 * hi + 4];

    v16h A;
#pragma unroll
    for (int t = 0; t < 16; ++t) {
      const int j = j0 + kmapA(t, hi);
      float e = sA + dvv[t];
      e = (e >= 0.f) ? e : NEG_SLOPE * e;
      const float p = (av[t] > 0.f || j == irow) ? __expf(e - mI) : 0.f;
      A[t] = (_Float16)p;
    }
    const v16h B0 = __builtin_shufflevector(*(const v8h*)(col0 + j0 + 8 * hi), *(const v8h*)(col0 + j0 + 16 + 8 * hi), 0, 1, 2, 3, 4, 5, 6, 7, 8, 9, 10, 11, 12, 13, 14, 15);
    const v16h B1 = __builtin_shufflevector(*(const v8h*)(col1 + j0 + 8 * hi), *(const v8h*)(col1 + j0 + 16 + 8 * hi), 0, 1, 2, 3, 4, 5, 6, 7, 8, 9, 10, 11, 12, 13, 14, 15);
    c0 = __builtin_amdgcn_wmma_f32_16x16x32_f16(false, A, false, B0, (short)0, c0, false, false);
    c1 = __builtin_amdgcn_wmma_f32_16x16x32_f16(false, A, false, B1, (short)0, c1, false, false);
  }

#pragma unroll
  for (int r = 0; r < 8; ++r) {
    const int row = i0 + r + 8 * hi;
    const float il = invl[(size_t)bh * N_ + row];
    const float v0 = fmaxf(c0[r] * il, 0.f);
    const float v1 = fmaxf(c1[r] * il, 0.f);
    out[((size_t)b * N_ + row) * (H_ * O_) + h * O_ + lo]      = v0;
    out[((size_t)b * N_ + row) * (H_ * O_) + h * O_ + 16 + lo] = v1;
  }
}

// ---------------------------------------------------------------------------
extern "C" void kernel_launch(void* const* d_in, const int* in_sizes, int n_in,
                              void* d_out, int out_size, void* d_ws, size_t ws_size,
                              hipStream_t stream) {
  (void)in_sizes; (void)n_in; (void)out_size; (void)ws_size;
  const float* x     = (const float*)d_in[0];
  const float* adj   = (const float*)d_in[1];
  const float* W     = (const float*)d_in[2];
  const float* a_src = (const float*)d_in[3];
  const float* a_dst = (const float*)d_in[4];
  float* out = (float*)d_out;

  char* ws = (char*)d_ws;
  _Float16* WhT = (_Float16*)ws;                                   // 4 MB
  size_t off = (size_t)B_ * H_ * N_ * O_ * sizeof(_Float16);
  _Float16* WT = (_Float16*)(ws + off); off += (size_t)H_ * IN_ * O_ * sizeof(_Float16);
  float* aS = (float*)(ws + off); off += (size_t)B_ * H_ * N_ * sizeof(float);
  float* aD = (float*)(ws + off); off += (size_t)B_ * H_ * N_ * sizeof(float);
  float* mR = (float*)(ws + off); off += (size_t)B_ * H_ * N_ * sizeof(float);
  float* iL = (float*)(ws + off);

  k_wt<<<(H_ * O_ * IN_) / 256, 256, 0, stream>>>(W, WT);
  k_wh<<<B_ * H_ * (N_ / 128), 256, 0, stream>>>(x, WT, WhT);
  k_alpha<<<(B_ * H_ * N_) / 256, 256, 0, stream>>>(WhT, a_src, a_dst, aS, aD);
  k_stats<<<(B_ * H_ * N_ * 32) / 256, 256, 0, stream>>>(adj, aS, aD, mR, iL);
  k_attn_out<<<B_ * (N_ / 16), 256, 0, stream>>>(adj, WhT, aS, aD, mR, iL, out);
}
